// LlamaRoPEAttention_31430570672331
// MI455X (gfx1250) — matrix operand order repaired and checked by reading, not compiled, not run on hardware
//
#include <hip/hip_runtime.h>
#include <stdint.h>

// ---------------- problem constants (from reference) ----------------
#define HIDDEN 2048
#define NHEAD  16
#define HDIM   128          // HIDDEN / NHEAD
#define SEQ    2048
#define BATCH  2
#define MROWS  (BATCH * SEQ)   // 4096 rows of x / q / k / v

// ---------------- CDNA5 WMMA / TDM types ----------------
typedef __bf16 v16bf __attribute__((ext_vector_type(16)));
typedef float  v8f   __attribute__((ext_vector_type(8)));
typedef unsigned int u32x4 __attribute__((ext_vector_type(4)));
typedef int  i32x8 __attribute__((ext_vector_type(8)));
typedef int  i32x4 __attribute__((ext_vector_type(4)));

union Frag { uint4 u[2]; v16bf v; };        // 32B: one bf16 A/B fragment per lane
union Acc  { float f[8]; v8f  v; };         // 32B: one f32 16x16 C/D tile per lane

__device__ __forceinline__ uint16_t f32_to_bf16(float f) {
  union { float f; uint32_t u; } x; x.f = f;
  uint32_t u = x.u;
  u += 0x7FFFu + ((u >> 16) & 1u);          // round-to-nearest-even
  return (uint16_t)(u >> 16);
}
__device__ __forceinline__ float bf16_to_f32(uint16_t h) {
  union { uint32_t u; float f; } x; x.u = ((uint32_t)h) << 16;
  return x.f;
}

#define WMMA_BF16(A, B, C) \
  __builtin_amdgcn_wmma_f32_16x16x32_bf16(false, (A), false, (B), (short)0, (C), false, false)

// ---------------------------------------------------------------------
// Tensor Data Mover: 2D tile load (global -> LDS) with row padding.
// Descriptor layout per CDNA5 ISA §8 (D# groups 0..3); groups 2/3 zero
// for a 2D tensor. data_size=1 (2-byte bf16). pad_interval/pad_amount
// insert LDS padding per row so ds_load_b128 fragment reads don't bank
// conflict. 6-arg builtin form (clang-23 / therock-10.0 headers).
// ---------------------------------------------------------------------
__device__ __forceinline__ void tdm_load_2d(uint32_t lds_off, const void* gsrc,
                                            uint32_t tile_d0, uint32_t tile_d1,
                                            uint32_t tensor_d0, uint32_t tensor_d1,
                                            uint32_t stride_elems,
                                            uint32_t pad_interval, uint32_t pad_amount) {
  const uint64_t ga = (uint64_t)(uintptr_t)gsrc;
  u32x4 g0;
  g0[0] = 1u;                                              // count=1 (valid), user mode
  g0[1] = lds_off;                                         // lds_addr [63:32]
  g0[2] = (uint32_t)ga;                                    // global_addr low
  g0[3] = (uint32_t)((ga >> 32) & 0x1FFFFFFu) | (2u << 30);// addr high | type=2
  i32x8 g1;
  g1[0] = (int)((1u << 16) | (1u << 20) |                  // data_size=2B, pad_enable
                (pad_interval << 22) | (pad_amount << 25));
  g1[1] = (int)(tensor_d0 << 16);                          // tensor_dim0[15:0]
  g1[2] = (int)((tensor_d0 >> 16) | (tensor_d1 << 16));    // dim0 hi | dim1 lo
  g1[3] = (int)((tensor_d1 >> 16) | (tile_d0 << 16));      // dim1 hi | tile_dim0
  g1[4] = (int)(tile_d1 & 0xFFFFu);                        // tile_dim1 (tile_dim2=0)
  g1[5] = (int)stride_elems;                               // tensor_dim0_stride lo
  g1[6] = 0;                                               // stride hi | dim1_stride lo
  g1[7] = 0;
  i32x4 gz4 = {0, 0, 0, 0};
  i32x8 gz8 = {0, 0, 0, 0, 0, 0, 0, 0};
  __builtin_amdgcn_tensor_load_to_lds(g0, g1, gz4, gz4, gz8, 0);
}

// =====================================================================
// Kernel 1: QKV projection  C[M,N] = X[M,K] * W[N,K]^T  (bf16 WMMA, f32 acc)
//   gridDim = (N/128, M/128, 3)  blockDim = 256 (8 waves)
//   z==0 -> Q, z==1 -> K (stored [row, HIDDEN] bf16)
//   z==2 -> V stored transposed as Vt[b, h, d, s] bf16
// =====================================================================
#define LDS_STRIDE 48   // u16 per row; 96B keeps b128 alignment, spreads banks

__global__ __launch_bounds__(256)
void qkv_gemm_wmma(const float* __restrict__ x,
                   const float* __restrict__ wq,
                   const float* __restrict__ wk,
                   const float* __restrict__ wv,
                   uint16_t* __restrict__ qb,
                   uint16_t* __restrict__ kb,
                   uint16_t* __restrict__ vtb) {
  __shared__ uint16_t lA[128 * LDS_STRIDE];
  __shared__ uint16_t lB[128 * LDS_STRIDE];

  const int z = blockIdx.z;
  const float* W = (z == 0) ? wq : (z == 1) ? wk : wv;

  const int nBase = blockIdx.x * 128;
  const int mBase = blockIdx.y * 128;

  const int t    = threadIdx.x;
  const int lane = t & 31;
  const int wave = t >> 5;
  const int half = lane >> 4;     // which 16-lane group
  const int lr   = lane & 15;
  const int wm   = wave >> 2;     // 0..1  (64-row slab)
  const int wn   = wave & 3;      // 0..3  (32-col slab)

  Acc acc[4][2] = {};

  // staging coordinates: 256 threads move a 128x32 f32 tile (each 4x float4)
  const int sr = t >> 3;          // 0..31
  const int sc = (t & 7) * 4;     // 0,4,...,28

  for (int ks = 0; ks < HIDDEN; ks += 32) {
    // ---- stage A (x) and B (W) tiles, converting f32 -> bf16 into LDS ----
    #pragma unroll
    for (int rr = 0; rr < 4; ++rr) {
      const int r = sr + rr * 32;
      const float4 av = *(const float4*)(x + (size_t)(mBase + r) * HIDDEN + ks + sc);
      const float4 bv = *(const float4*)(W + (size_t)(nBase + r) * HIDDEN + ks + sc);
      uint2 pa, pb;
      pa.x = (uint32_t)f32_to_bf16(av.x) | ((uint32_t)f32_to_bf16(av.y) << 16);
      pa.y = (uint32_t)f32_to_bf16(av.z) | ((uint32_t)f32_to_bf16(av.w) << 16);
      pb.x = (uint32_t)f32_to_bf16(bv.x) | ((uint32_t)f32_to_bf16(bv.y) << 16);
      pb.y = (uint32_t)f32_to_bf16(bv.z) | ((uint32_t)f32_to_bf16(bv.w) << 16);
      *(uint2*)&lA[r * LDS_STRIDE + sc] = pa;
      *(uint2*)&lB[r * LDS_STRIDE + sc] = pb;
    }
    __syncthreads();

    // ---- fragment loads: 2x ds_load_b128 each (ISA 7.12.2 16-bit layouts) ----
    Frag afr[4], bfr[2];
    #pragma unroll
    for (int a = 0; a < 4; ++a) {
      const int row = wm * 64 + a * 16 + lr;
      afr[a].u[0] = *(const uint4*)&lA[row * LDS_STRIDE + half * 8];
      afr[a].u[1] = *(const uint4*)&lA[row * LDS_STRIDE + 16 + half * 8];
    }
    #pragma unroll
    for (int b = 0; b < 2; ++b) {
      const int row = wn * 32 + b * 16 + lr;
      bfr[b].u[0] = *(const uint4*)&lB[row * LDS_STRIDE + half * 8];
      bfr[b].u[1] = *(const uint4*)&lB[row * LDS_STRIDE + 16 + half * 8];
    }
    #pragma unroll
    for (int a = 0; a < 4; ++a)
      #pragma unroll
      for (int b = 0; b < 2; ++b)
        acc[a][b].v = WMMA_BF16(afr[a].v, bfr[b].v, acc[a][b].v);
    __syncthreads();
  }

  // ---- epilogue (uniform branch hoisted out of the store loops) ----
  if (z != 2) {
    uint16_t* outp = (z == 0) ? qb : kb;
    #pragma unroll
    for (int a = 0; a < 4; ++a)
      #pragma unroll
      for (int b = 0; b < 2; ++b)
        #pragma unroll
        for (int i = 0; i < 8; ++i) {
          const int row = mBase + wm * 64 + a * 16 + i + 8 * half;
          const int col = nBase + wn * 32 + b * 16 + lr;
          outp[(size_t)row * HIDDEN + col] = f32_to_bf16(acc[a][b].f[i]);
        }
  } else {
    #pragma unroll
    for (int a = 0; a < 4; ++a)
      #pragma unroll
      for (int b = 0; b < 2; ++b)
        #pragma unroll
        for (int i = 0; i < 8; ++i) {
          const int row = mBase + wm * 64 + a * 16 + i + 8 * half;
          const int col = nBase + wn * 32 + b * 16 + lr;
          const int bb = row >> 11, ss = row & (SEQ - 1);   // SEQ = 2048
          const int h = col >> 7, d = col & (HDIM - 1);
          vtb[(((size_t)bb * NHEAD + h) * HDIM + d) * SEQ + ss] =
              f32_to_bf16(acc[a][b].f[i]);
        }
  }
}

// =====================================================================
// Kernel 2: RoPE in-place on Q and K (bf16)
// =====================================================================
__global__ __launch_bounds__(256)
void rope_kernel(uint16_t* __restrict__ qb, uint16_t* __restrict__ kb) {
  const size_t nPairs = (size_t)MROWS * NHEAD * (HDIM / 2);
  size_t idx = (size_t)blockIdx.x * blockDim.x + threadIdx.x;
  if (idx >= 2 * nPairs) return;
  uint16_t* buf = (idx < nPairs) ? qb : kb;
  size_t p = idx % nPairs;
  const int d = (int)(p % (HDIM / 2)); p /= (HDIM / 2);
  const int h = (int)(p % NHEAD);      p /= NHEAD;
  const int row = (int)p;              // 0..MROWS-1
  const int s = row % SEQ;
  // freq_d = theta^(-2d/HDIM) = exp(-d * ln(10000)/64)
  const float freq = __expf(-(float)d * 0.14391157f);
  float sn, cs;
  __sincosf((float)s * freq, &sn, &cs);
  const size_t base = (size_t)row * HIDDEN + (size_t)h * HDIM;
  const float x1 = bf16_to_f32(buf[base + d]);
  const float x2 = bf16_to_f32(buf[base + d + 64]);
  buf[base + d]      = f32_to_bf16(x1 * cs - x2 * sn);
  buf[base + d + 64] = f32_to_bf16(x2 * cs + x1 * sn);
}

// =====================================================================
// Kernel 3: flash attention (no causal mask, matches reference)
//   gridDim = (SEQ/64, NHEAD, BATCH), blockDim = 128 (4 waves)
//   Each wave owns a 16-row Q tile. Per 32-key step the block stages the
//   shared K tile (32x128) and V^T tile (128x32) into LDS ONCE via the
//   Tensor Data Mover (TDM), with row padding so fragment ds_load_b128
//   reads are conflict-free. Wave 0 issues the TDM descriptors and waits
//   on TENSORcnt; a workgroup barrier publishes the tiles.
// =====================================================================
#define KT_STRIDE 136   // u16: 128 elems + 8 pad  (272 B rows -> 4-bank rotation)
#define VT_STRIDE 40    // u16: 32 elems + 8 pad   (80 B rows)

__global__ __launch_bounds__(128)
void flash_attn_wmma(const uint16_t* __restrict__ Q,
                     const uint16_t* __restrict__ K,
                     const uint16_t* __restrict__ Vt,
                     float* __restrict__ out) {
  __shared__ uint16_t kTile[32 * KT_STRIDE];    // 8704 B
  __shared__ uint16_t vTile[128 * VT_STRIDE];   // 10240 B
  __shared__ uint16_t Pbuf[4][16 * LDS_STRIDE]; // 6144 B

  const int lane = threadIdx.x & 31;
  const int wave = threadIdx.x >> 5;
  const int half = lane >> 4;
  const int lr   = lane & 15;
  const int b  = blockIdx.z;
  const int h  = blockIdx.y;
  const int q0 = blockIdx.x * 64 + wave * 16;
  const float scale = 0.088388347762f;   // 1/sqrt(128)

  const uint16_t* kbase = K  + ((size_t)b * SEQ) * HIDDEN + h * HDIM;
  const uint16_t* vbase = Vt + ((size_t)(b * NHEAD + h)) * HDIM * SEQ;

  // ---- preload Q fragments (A layout, 4 x 32-wide d chunks) ----
  Frag qf[4];
  const uint16_t* qrow = Q + ((size_t)(b * SEQ + q0 + lr)) * HIDDEN + h * HDIM;
  #pragma unroll
  for (int kc = 0; kc < 4; ++kc) {
    qf[kc].u[0] = *(const uint4*)(qrow + kc * 32 + half * 8);
    qf[kc].u[1] = *(const uint4*)(qrow + kc * 32 + 16 + half * 8);
  }

  Acc o[8] = {};
  float mrow[8], lsum[8];
  #pragma unroll
  for (int i = 0; i < 8; ++i) { mrow[i] = -1e30f; lsum[i] = 0.0f; }

  const uint32_t kTileOff = (uint32_t)(uintptr_t)&kTile[0];
  const uint32_t vTileOff = (uint32_t)(uintptr_t)&vTile[0];
  uint16_t* P = &Pbuf[wave][0];

  for (int kt = 0; kt < SEQ; kt += 32) {
    // ---- TDM: stage K (32 keys x 128 d) and V^T (128 d x 32 keys) ----
    if (wave == 0) {
      // K rows: 128 elems = 64 DW -> pad_interval=5 (64 DW), pad_amount=3 (4 DW)
      tdm_load_2d(kTileOff, kbase + (size_t)kt * HIDDEN,
                  /*tile*/ 128, 32, /*tensor*/ HIDDEN, SEQ,
                  /*stride*/ HIDDEN, /*pad_i*/ 5, /*pad_a*/ 3);
      // V rows: 32 elems = 16 DW -> pad_interval=3 (16 DW), pad_amount=3 (4 DW)
      tdm_load_2d(vTileOff, vbase + kt,
                  /*tile*/ 32, 128, /*tensor*/ SEQ, HDIM,
                  /*stride*/ SEQ, /*pad_i*/ 3, /*pad_a*/ 3);
      __builtin_amdgcn_s_wait_tensorcnt(0);
    } else if (wave == 1 && kt + 32 < SEQ) {
      __builtin_prefetch(kbase + (size_t)(kt + 32 + lane) * HIDDEN, 0, 1);
    }
    __syncthreads();

    // ---- S = Q K^T for two 16-key tiles (K fragments from LDS) ----
    Acc s0 = {}, s1 = {};
    #pragma unroll
    for (int kc = 0; kc < 4; ++kc) {
      Frag k0f, k1f;
      const uint16_t* kr0 = &kTile[(size_t)lr * KT_STRIDE + kc * 32 + half * 8];
      const uint16_t* kr1 = &kTile[(size_t)(16 + lr) * KT_STRIDE + kc * 32 + half * 8];
      k0f.u[0] = *(const uint4*)(kr0);
      k0f.u[1] = *(const uint4*)(kr0 + 16);
      k1f.u[0] = *(const uint4*)(kr1);
      k1f.u[1] = *(const uint4*)(kr1 + 16);
      s0.v = WMMA_BF16(qf[kc].v, k0f.v, s0.v);
      s1.v = WMMA_BF16(qf[kc].v, k1f.v, s1.v);
    }

    // ---- online softmax (row stats live in lane groups of 16) ----
    #pragma unroll
    for (int i = 0; i < 8; ++i) {
      const float v0 = s0.f[i] * scale;
      const float v1 = s1.f[i] * scale;
      float t = fmaxf(v0, v1);
      #pragma unroll
      for (int m = 1; m < 16; m <<= 1) t = fmaxf(t, __shfl_xor(t, m, 32));
      const float mn   = fmaxf(mrow[i], t);
      const float corr = __expf(mrow[i] - mn);
      const float p0 = __expf(v0 - mn);
      const float p1 = __expf(v1 - mn);
      float rs = p0 + p1;
      #pragma unroll
      for (int m = 1; m < 16; m <<= 1) rs += __shfl_xor(rs, m, 32);
      lsum[i] = lsum[i] * corr + rs;
      mrow[i] = mn;
      #pragma unroll
      for (int dt = 0; dt < 8; ++dt) o[dt].f[i] *= corr;
      const int prow = i + 8 * half;             // C-layout row
      P[prow * LDS_STRIDE + lr]      = f32_to_bf16(p0);
      P[prow * LDS_STRIDE + 16 + lr] = f32_to_bf16(p1);
    }
    asm volatile("s_wait_dscnt 0" ::: "memory");  // P store->load, same wave

    // ---- O += P V : P as A fragment, V^T rows as B fragments (LDS) ----
    Frag pf;
    pf.u[0] = *(const uint4*)&P[lr * LDS_STRIDE + half * 8];
    pf.u[1] = *(const uint4*)&P[lr * LDS_STRIDE + 16 + half * 8];
    #pragma unroll
    for (int dt = 0; dt < 8; ++dt) {
      Frag vf;
      const uint16_t* vr = &vTile[(size_t)(dt * 16 + lr) * VT_STRIDE + half * 8];
      vf.u[0] = *(const uint4*)(vr);
      vf.u[1] = *(const uint4*)(vr + 16);
      o[dt].v = WMMA_BF16(pf.v, vf.v, o[dt].v);
    }
    __syncthreads();   // all waves done with kTile/vTile before next TDM
  }

  // ---- normalize and store f32 output [B, S, HIDDEN] ----
  float* obase = out + ((size_t)(b * SEQ + q0)) * HIDDEN + h * HDIM;
  #pragma unroll
  for (int i = 0; i < 8; ++i) {
    const float inv = 1.0f / lsum[i];
    const int row = i + 8 * half;
    #pragma unroll
    for (int dt = 0; dt < 8; ++dt)
      obase[(size_t)row * HIDDEN + dt * 16 + lr] = o[dt].f[i] * inv;
  }
}

// =====================================================================
extern "C" void kernel_launch(void* const* d_in, const int* in_sizes, int n_in,
                              void* d_out, int out_size, void* d_ws, size_t ws_size,
                              hipStream_t stream) {
  const float* x  = (const float*)d_in[0];
  const float* wq = (const float*)d_in[1];
  const float* wk = (const float*)d_in[2];
  const float* wv = (const float*)d_in[3];
  float* out = (float*)d_out;

  // workspace: Q, K (bf16 [MROWS, HIDDEN]) and Vt (bf16 [B, H, D, S]) = 48 MB
  const size_t elems = (size_t)MROWS * HIDDEN;
  uint16_t* qb  = (uint16_t*)d_ws;
  uint16_t* kb  = qb + elems;
  uint16_t* vtb = kb + elems;

  qkv_gemm_wmma<<<dim3(HIDDEN / 128, MROWS / 128, 3), 256, 0, stream>>>(
      x, wq, wk, wv, qb, kb, vtb);

  const size_t nPairs = (size_t)MROWS * NHEAD * (HDIM / 2) * 2;  // Q and K
  rope_kernel<<<(unsigned)((nPairs + 255) / 256), 256, 0, stream>>>(qb, kb);

  flash_attn_wmma<<<dim3(SEQ / 64, NHEAD, BATCH), 128, 0, stream>>>(
      qb, kb, vtb, out);
}
